// TokenInfoNCE_17308718202928
// MI455X (gfx1250) — matrix operand order repaired and checked by reading, not compiled, not run on hardware
//
#include <hip/hip_runtime.h>
#include <hip/hip_bf16.h>

typedef __attribute__((ext_vector_type(16))) __bf16 v16bf;
typedef __attribute__((ext_vector_type(8)))  __bf16 v8bf;
typedef __attribute__((ext_vector_type(8)))  float  v8f;

#define DIM    768
#define BATCH  512
#define NCOLS  8192          // BATCH * P
#define INV_T  10.0f         // 1 / 0.1

__device__ __forceinline__ unsigned short f32_to_bf16_rne(float f) {
  unsigned u = __float_as_uint(f);
  u += 0x7FFFu + ((u >> 16) & 1u);
  return (unsigned short)(u >> 16);
}

// ---------------------------------------------------------------------------
// Kernel 1: L2-normalize rows of global_feat (512xD) and part_feat (8192xD),
// emit bf16. Block 0 also zeroes the per-row sum accumulators (1024 floats).
// ---------------------------------------------------------------------------
__global__ __launch_bounds__(256)
void nrm_bf16_kernel(const float* __restrict__ gf, const float* __restrict__ pf,
                     unsigned short* __restrict__ Gn, unsigned short* __restrict__ Pn,
                     float* __restrict__ rowAcc /* 1024 floats: all|pos */) {
  const int row = blockIdx.x;
  const int t   = threadIdx.x;

  const float* src;
  unsigned short* dst;
  if (row < BATCH) { src = gf + (size_t)row * DIM;          dst = Gn + (size_t)row * DIM; }
  else             { src = pf + (size_t)(row - BATCH) * DIM; dst = Pn + (size_t)(row - BATCH) * DIM; }

  float v[3];
  float ss = 0.0f;
#pragma unroll
  for (int i = 0; i < 3; ++i) { v[i] = src[t + i * 256]; ss += v[i] * v[i]; }

#pragma unroll
  for (int off = 16; off; off >>= 1) ss += __shfl_xor(ss, off);

  __shared__ float wsum[8];
  const int wv = t >> 5, ln = t & 31;
  if (ln == 0) wsum[wv] = ss;
  __syncthreads();
  float tot = 0.0f;
#pragma unroll
  for (int i = 0; i < 8; ++i) tot += wsum[i];
  const float scale = rsqrtf(tot);

#pragma unroll
  for (int i = 0; i < 3; ++i) dst[t + i * 256] = f32_to_bf16_rne(v[i] * scale);

  if (row == 0) {
    rowAcc[t]       = 0.0f;
    rowAcc[t + 256] = 0.0f;
    rowAcc[t + 512] = 0.0f;
    rowAcc[t + 768] = 0.0f;
  }
}

// ---------------------------------------------------------------------------
// Kernel 2: fused bf16 WMMA GEMM (C = Gn * Pn^T, 512x8192, K=768) with
// streaming per-row sum(exp(C/T)) for "all" and "pos" (label-matched) columns.
// Grid: (32 M-tiles, 16 column groups); 8 waves/block, each wave computes
// FOUR 16x16 column tiles in a single K pass so one A-fragment load feeds
// four v_wmma instructions (loads/wmma = 2.5 instead of 4).
// ---------------------------------------------------------------------------
__global__ __launch_bounds__(256)
void gemm_lse_kernel(const unsigned short* __restrict__ Gnu,
                     const unsigned short* __restrict__ Pnu,
                     const int* __restrict__ labels,
                     float* __restrict__ rowAll, float* __restrict__ rowPos) {
  const int tid  = threadIdx.x;
  const int wave = tid >> 5;
  const int lane = tid & 31;
  const int l16  = lane & 15;
  const int half = lane >> 4;        // 0: rows M=0..7 of C, 1: rows M=8..15

  const int b0  = blockIdx.x * 16;                 // row-tile base (rows of Gn)
  const int jt0 = blockIdx.y * 32 + wave * 4;      // first of 4 column tiles

  const __bf16* Amat = reinterpret_cast<const __bf16*>(Gnu);
  const __bf16* Bmat = reinterpret_cast<const __bf16*>(Pnu);

  const __bf16* aRow = Amat + (size_t)(b0 + l16) * DIM;

  const __bf16* bRow[4];
  int labCol[4];
#pragma unroll
  for (int t = 0; t < 4; ++t) {
    const int jt = jt0 + t;
    bRow[t]   = Bmat + (size_t)(jt * 16 + l16) * DIM;
    labCol[t] = labels[jt];                        // uniform over the 16 columns
  }

  int rowLab[8];
#pragma unroll
  for (int r = 0; r < 8; ++r) rowLab[r] = labels[b0 + half * 8 + r];

  v8f c0 = {0,0,0,0,0,0,0,0};
  v8f c1 = {0,0,0,0,0,0,0,0};
  v8f c2 = {0,0,0,0,0,0,0,0};
  v8f c3 = {0,0,0,0,0,0,0,0};

#pragma unroll 2
  for (int k0 = 0; k0 < DIM; k0 += 32) {
    // A fragment: lanes 0-15 hold K = k0+{0..7,16..23}; lanes 16-31 hold +8
    v8bf a_lo = *reinterpret_cast<const v8bf*>(aRow + k0 + half * 8);
    v8bf a_hi = *reinterpret_cast<const v8bf*>(aRow + k0 + half * 8 + 16);
    v16bf a;
#pragma unroll
    for (int e = 0; e < 8; ++e) { a[e] = a_lo[e]; a[e + 8] = a_hi[e]; }

    v8f* acc[4] = {&c0, &c1, &c2, &c3};
#pragma unroll
    for (int t = 0; t < 4; ++t) {
      // B fragment (N-major rows of Pn): lanes 0-15 K=k0+{0..7,16..23}, lanes 16-31 +8
      v8bf b_lo = *reinterpret_cast<const v8bf*>(bRow[t] + k0 + half * 8);
      v8bf b_hi = *reinterpret_cast<const v8bf*>(bRow[t] + k0 + half * 8 + 16);
      v16bf b;
#pragma unroll
      for (int e = 0; e < 8; ++e) { b[e] = b_lo[e]; b[e + 8] = b_hi[e]; }
      *acc[t] = __builtin_amdgcn_wmma_f32_16x16x32_bf16(
                    /*neg_a=*/false, a, /*neg_b=*/false, b,
                    /*c_mod=*/(short)0, *acc[t],
                    /*reuse_a=*/false, /*reuse_b=*/false);
    }
  }

  // epilogue: exp(C/T), accumulate per (row, this lane's column)
  float sAll[8], sPos[8];
#pragma unroll
  for (int r = 0; r < 8; ++r) { sAll[r] = 0.0f; sPos[r] = 0.0f; }

  const v8f* accv[4] = {&c0, &c1, &c2, &c3};
#pragma unroll
  for (int t = 0; t < 4; ++t) {
#pragma unroll
    for (int r = 0; r < 8; ++r) {
      const float e = __expf((*accv[t])[r] * INV_T);
      sAll[r] += e;
      if (rowLab[r] == labCol[t]) sPos[r] += e;
    }
  }

  // reduce over the 16 columns (lanes within each half-group); xor<=8 stays
  // inside the 16-lane group, so both halves reduce independently.
#pragma unroll
  for (int r = 0; r < 8; ++r) {
#pragma unroll
    for (int off = 8; off; off >>= 1) {
      sAll[r] += __shfl_xor(sAll[r], off);
      sPos[r] += __shfl_xor(sPos[r], off);
    }
  }

  __shared__ float shAll[16], shPos[16];
  if (tid < 16) { shAll[tid] = 0.0f; shPos[tid] = 0.0f; }
  __syncthreads();
  if (l16 == 0) {   // lane 0 -> rows 0..7, lane 16 -> rows 8..15
#pragma unroll
    for (int r = 0; r < 8; ++r) {
      atomicAdd(&shAll[half * 8 + r], sAll[r]);
      atomicAdd(&shPos[half * 8 + r], sPos[r]);
    }
  }
  __syncthreads();
  if (tid < 16) {
    atomicAdd(&rowAll[b0 + tid], shAll[tid]);
    atomicAdd(&rowPos[b0 + tid], shPos[tid]);
  }
}

// ---------------------------------------------------------------------------
// Kernel 3: result = mean_b( log(sumAll_b) - log(sumPos_b) )
// ---------------------------------------------------------------------------
__global__ __launch_bounds__(512)
void finalize_kernel(const float* __restrict__ rowAll,
                     const float* __restrict__ rowPos,
                     float* __restrict__ out) {
  const int t = threadIdx.x;  // 512 threads, one per distinct row
  float v = __logf(rowAll[t]) - __logf(rowPos[t]);
#pragma unroll
  for (int off = 16; off; off >>= 1) v += __shfl_xor(v, off);

  __shared__ float wsum[16];
  const int wv = t >> 5, ln = t & 31;
  if (ln == 0) wsum[wv] = v;
  __syncthreads();
  if (t == 0) {
    float tot = 0.0f;
#pragma unroll
    for (int i = 0; i < 16; ++i) tot += wsum[i];
    out[0] = tot * (1.0f / 512.0f);
  }
}

// ---------------------------------------------------------------------------
extern "C" void kernel_launch(void* const* d_in, const int* in_sizes, int n_in,
                              void* d_out, int out_size, void* d_ws, size_t ws_size,
                              hipStream_t stream) {
  const float* gf     = (const float*)d_in[0];   // [512, 768] f32
  const float* pf     = (const float*)d_in[1];   // [512, 16, 768] f32
  const int*   labels = (const int*)d_in[2];     // [512]

  char* ws = (char*)d_ws;
  unsigned short* Gn = (unsigned short*)ws;                                   // 512*768 bf16
  unsigned short* Pn = (unsigned short*)(ws + (size_t)BATCH * DIM * 2);       // 8192*768 bf16
  float* rowAll = (float*)(ws + (size_t)BATCH * DIM * 2 + (size_t)NCOLS * DIM * 2);
  float* rowPos = rowAll + BATCH;

  nrm_bf16_kernel<<<BATCH + NCOLS, 256, 0, stream>>>(gf, pf, Gn, Pn, rowAll);

  dim3 grid(BATCH / 16, NCOLS / 16 / 32);  // (32, 16)
  gemm_lse_kernel<<<grid, 256, 0, stream>>>(Gn, Pn, labels, rowAll, rowPos);

  finalize_kernel<<<1, 512, 0, stream>>>(rowAll, rowPos, (float*)d_out);
}
